// VectorQuantizer_70016556859795
// MI455X (gfx1250) — compile-verified, matrix operand order checked by reading, not run on hardware
//
#include <hip/hip_runtime.h>
#include <hip/hip_bf16.h>
#include <stdint.h>

// Problem constants (from reference)
#define KCB   1024      // codebook size
#define DCH   256       // code dim / channels
#define TT    4096      // time steps
#define BB    16        // batch

typedef __attribute__((ext_vector_type(16))) __bf16   v16bf;
typedef __attribute__((ext_vector_type(8)))  float    v8f;
typedef __attribute__((ext_vector_type(4)))  uint32_t u32x4;
typedef __attribute__((ext_vector_type(8)))  uint32_t u32x8;

// Dynamic LDS layout (bytes)
#define L_BUF0   0          // 8 KB: B tile double-buffer 0
#define L_BUF1   8192       // 8 KB: B tile double-buffer 1
#define L_NORMS  16384      // 4 KB: ||e||^2
#define L_SIDX   20480      // 8 waves x 32 rows x int
#define L_SLOSS  21504      // 8 floats
#define L_TOTAL  21536

// ---------------------------------------------------------------------------
// Prep: pack codebook into WMMA-B fragment layout (bf16) + row norms ||e||^2.
// wsB dword index = ((nTile*8 + kk)*32 + lane)*8 + r  -> one 8KB tile per nTile
// ---------------------------------------------------------------------------
__global__ __launch_bounds__(256) void vq_prep(const float* __restrict__ cb,
                                               uint32_t* __restrict__ wsB,
                                               float* __restrict__ norms) {
  int idx = blockIdx.x * 256 + threadIdx.x;            // 0..131071
  int r    = idx & 7;
  int lane = (idx >> 3) & 31;
  int tile = idx >> 8;                                  // nt*8 + kk
  int kk = tile & 7, nt = tile >> 3;
  int n  = nt * 16 + (lane & 15);
  int c0 = kk * 32 + (lane >> 4) * 16 + 2 * r;
  union { __bf16 h[2]; uint32_t u; } p;
  p.h[0] = (__bf16)cb[n * DCH + c0];
  p.h[1] = (__bf16)cb[n * DCH + c0 + 1];
  wsB[idx] = p.u;
  if (idx < KCB) {
    const float* row = cb + (long)idx * DCH;
    float s = 0.f;
    for (int c = 0; c < DCH; ++c) s += row[c] * row[c];
    norms[idx] = s;
  }
}

// ---------------------------------------------------------------------------
// Main: 8 waves/WG, 32 rows/wave (two A panels). B tiles staged to LDS by the
// Tensor Data Mover (double-buffered, one issuing wave), consumed by all 8
// waves via ds_load_b128 -> v_wmma_f32_16x16x32_bf16.
// ---------------------------------------------------------------------------
__global__ __launch_bounds__(256) void vq_main(const float* __restrict__ x,
                                               const float* __restrict__ cb,
                                               const uint32_t* __restrict__ wsB,
                                               const float* __restrict__ norms,
                                               float* __restrict__ qOut,
                                               float* __restrict__ idxOut,
                                               float* __restrict__ partials) {
  extern __shared__ char smem[];

  const int lane = threadIdx.x & 31;
  const int wave = threadIdx.x >> 5;
  const int half = lane >> 4;
  const int l    = lane & 15;

  // 256 rows per WG, 32 per wave (two 16-row panels)
  const long rowBase = (long)blockIdx.x * 256 + wave * 32;
  const int  bb = (int)(rowBase >> 12);
  const int  t0 = (int)(rowBase & (TT - 1));
  const float* xb = x + (long)bb * DCH * TT + t0;       // + c*TT + t_local

  // ---- stage norms into LDS (all threads) ----
  ((float4*)(smem + L_NORMS))[threadIdx.x] = ((const float4*)norms)[threadIdx.x];

  // ---- TDM prologue: wave 0 DMAs code-tile 0 into BUF0 ----
  const uint64_t wsBase  = (uint64_t)(uintptr_t)wsB;
  const uint32_t ldsBase = (uint32_t)(uintptr_t)smem;   // generic->LDS: low 32 bits
  if (wave == 0) {
    u32x4 g0; u32x8 g1;
    g0[0] = 1u;                                  // count=1, no gather
    g0[1] = ldsBase + L_BUF0;                    // lds_addr
    g0[2] = (uint32_t)wsBase;                    // global_addr[31:0]
    g0[3] = 0x80000000u | (uint32_t)(wsBase >> 32);  // type=2 | addr[56:32]
    g1[0] = 0x00020000u;                         // data_size=4B, mask=0
    g1[1] = 0x08000000u;                         // tensor_dim0 = 2048 (lo16 in [63:48])
    g1[2] = 0x00010000u;                         // tensor_dim1 = 1
    g1[3] = 0x08000000u;                         // tile_dim0 = 2048
    g1[4] = 0x00000001u;                         // tile_dim1 = 1
    g1[5] = 2048u;                               // tensor_dim0_stride = 2048
    g1[6] = 0u; g1[7] = 0u;
    asm volatile("tensor_load_to_lds %0, %1" :: "s"(g0), "s"(g1) : "memory");
  }

  // ---- Load two A panels (16 rows x 256 ch each) as bf16 fragments ----
  v16bf a0[8], a1[8];
#pragma unroll
  for (int kk = 0; kk < 8; ++kk) {
#pragma unroll
    for (int r = 0; r < 8; ++r) {
      const int c0 = kk * 32 + half * 16 + 2 * r;
      const float* p0 = xb + (long)c0 * TT;
      const float* p1 = p0 + TT;
      a0[kk][2*r]   = (__bf16)p0[l];
      a0[kk][2*r+1] = (__bf16)p1[l];
      a1[kk][2*r]   = (__bf16)p0[16 + l];
      a1[kk][2*r+1] = (__bf16)p1[16 + l];
    }
  }

  float minv[16];
  int   mini[16];
#pragma unroll
  for (int r = 0; r < 16; ++r) { minv[r] = 3.4e38f; mini[r] = 0; }

  // ---- 64 code tiles; TDM double-buffer; 16 WMMAs per tile per wave ----
  for (int nt = 0; nt < 64; ++nt) {
    const int cur = nt & 1;
    if (wave == 0) __builtin_amdgcn_s_wait_tensorcnt((short)0);
    __syncthreads();                              // buf[cur] ready; prev reads done
    if (wave == 0 && nt + 1 < 64) {               // prefetch next tile
      const uint64_t ga = wsBase + (uint64_t)(nt + 1) * 8192u;
      u32x4 g0; u32x8 g1;
      g0[0] = 1u;
      g0[1] = ldsBase + (cur ? L_BUF0 : L_BUF1);
      g0[2] = (uint32_t)ga;
      g0[3] = 0x80000000u | (uint32_t)(ga >> 32);
      g1[0] = 0x00020000u; g1[1] = 0x08000000u; g1[2] = 0x00010000u;
      g1[3] = 0x08000000u; g1[4] = 0x00000001u; g1[5] = 2048u;
      g1[6] = 0u; g1[7] = 0u;
      asm volatile("tensor_load_to_lds %0, %1" :: "s"(g0), "s"(g1) : "memory");
    }

    const char* bbase = smem + (cur ? L_BUF1 : L_BUF0) + lane * 32;
    v8f acc0 = {}, acc1 = {};
#pragma unroll
    for (int kk = 0; kk < 8; ++kk) {
      union { uint4 q[2]; v16bf v; } bf;
      const uint4* p = (const uint4*)(bbase + kk * 1024);
      bf.q[0] = p[0];                             // ds_load_b128
      bf.q[1] = p[1];                             // ds_load_b128
      acc0 = __builtin_amdgcn_wmma_f32_16x16x32_bf16(
          false, a0[kk], false, bf.v, (short)0, acc0, false, false);
      acc1 = __builtin_amdgcn_wmma_f32_16x16x32_bf16(
          false, a1[kk], false, bf.v, (short)0, acc1, false, false);
    }

    const float nrm  = *(const float*)(smem + L_NORMS + (nt * 16 + l) * 4);
    const int   nIdx = nt * 16 + l;
#pragma unroll
    for (int r = 0; r < 8; ++r) {
      float d0 = __builtin_fmaf(-2.0f, acc0[r], nrm);
      float d1 = __builtin_fmaf(-2.0f, acc1[r], nrm);
      if (d0 < minv[r])     { minv[r]     = d0; mini[r]     = nIdx; }
      if (d1 < minv[8 + r]) { minv[8 + r] = d1; mini[8 + r] = nIdx; }
    }
  }

  // ---- argmin reduce across the 16 lanes of each half-wave ----
  int* sIdx = (int*)(smem + L_SIDX);
#pragma unroll
  for (int r = 0; r < 16; ++r) {
    float v = minv[r];
    int   i = mini[r];
#pragma unroll
    for (int off = 1; off < 16; off <<= 1) {
      float v2 = __shfl_xor(v, off, 32);
      int   i2 = __shfl_xor(i, off, 32);
      if (v2 < v || (v2 == v && i2 < i)) { v = v2; i = i2; }
    }
    // panel = r/8: rows panel*16 + half*8 + (r&7)
    if (l == 0) sIdx[wave * 32 + (r >> 3) * 16 + half * 8 + (r & 7)] = i;
  }
  __syncthreads();

  // ---- outputs: indices, quantized (transposed coalesced), fused loss ----
  float lossAcc = 0.f;
#pragma unroll
  for (int p = 0; p < 2; ++p) {
    const int myIdx = sIdx[wave * 32 + p * 16 + l];
    if (half == 0) idxOut[rowBase + p * 16 + l] = (float)myIdx;
    const float* myCb = cb + (long)myIdx * DCH;
    const long outBase = (long)bb * DCH * TT + t0 + p * 16 + l;
    const float* xbp = xb + p * 16;
#pragma unroll 4
    for (int cBase = 0; cBase < DCH; cBase += 2) {
      const int c = cBase + half;                 // 2 channels x 16 t per iter
      float q  = myCb[c];
      float xv = xbp[(long)c * TT + l];
      float df = q - xv;
      lossAcc = __builtin_fmaf(df, df, lossAcc);
      qOut[outBase + (long)c * TT] = q;
    }
  }

  // ---- loss: wave -> block -> ws partial (deterministic) ----
#pragma unroll
  for (int off = 1; off < 32; off <<= 1) lossAcc += __shfl_xor(lossAcc, off, 32);
  float* sLoss = (float*)(smem + L_SLOSS);
  if (lane == 0) sLoss[wave] = lossAcc;
  __syncthreads();
  if (threadIdx.x == 0) {
    float s = 0.f;
#pragma unroll
    for (int w = 0; w < 8; ++w) s += sLoss[w];
    partials[blockIdx.x] = s;
  }
}

// ---------------------------------------------------------------------------
// Final: reduce 256 partials -> loss = 1.25 * sum / (B*C*T)
// ---------------------------------------------------------------------------
__global__ __launch_bounds__(256) void vq_finish(const float* __restrict__ partials,
                                                 float* __restrict__ lossOut) {
  __shared__ float s[256];
  s[threadIdx.x] = partials[threadIdx.x];
  __syncthreads();
  for (int st = 128; st > 0; st >>= 1) {
    if (threadIdx.x < st) s[threadIdx.x] += s[threadIdx.x + st];
    __syncthreads();
  }
  if (threadIdx.x == 0)
    *lossOut = s[0] * (1.25f / ((float)BB * (float)DCH * (float)TT));
}

extern "C" void kernel_launch(void* const* d_in, const int* in_sizes, int n_in,
                              void* d_out, int out_size, void* d_ws, size_t ws_size,
                              hipStream_t stream) {
  const float* x  = (const float*)d_in[0];   // (16, 256, 4096) f32
  const float* cb = (const float*)d_in[1];   // (1024, 256) f32
  float* out = (float*)d_out;                // [quantized | indices | loss]

  uint32_t* wsB    = (uint32_t*)d_ws;                      // 512 KB packed bf16 B frags
  float*    norms  = (float*)((char*)d_ws + 131072 * 4);   // 4 KB
  float*    partls = norms + KCB;                          // 1 KB (256 partials)

  const long QOFF   = (long)BB * DCH * TT;   // 16777216
  const long IDXOFF = QOFF + (long)BB * TT;  // +65536

  vq_prep  <<<512, 256, 0, stream>>>(cb, wsB, norms);
  vq_main  <<<256, 256, L_TOTAL, stream>>>(x, cb, wsB, norms, out, out + QOFF, partls);
  vq_finish<<<1,   256, 0, stream>>>(partls, out + IDXOFF);
}
